// FlowForecastModel_66821101191594
// MI455X (gfx1250) — compile-verified, matrix operand order checked by reading, not run on hardware
//
#include <hip/hip_runtime.h>
#include <hip/hip_bf16.h>
#include <hip/hip_fp16.h>

#define NN  10000
#define NE  160000
#define SEQ 16
#define FIN 4
#define H1  64
#define H2  128
#define HOR 3

typedef __attribute__((ext_vector_type(16))) _Float16 v16h;
typedef __attribute__((ext_vector_type(8)))  float    v8f;

// ---------------------------------------------------------------------------
// WMMA fragment handling (CDNA5 wave32 layouts, CDNA5 ISA §7.12.2)
// A: 16x32 f16, lane<16 holds row M=lane, K {kb+0..7, kb+16..23};
//    lane>=16 holds row M=lane-16, K {kb+8..15, kb+24..31}
// B matrices are pre-swizzled into fragment-major layout so each lane reads
// 16 contiguous halfs (2x global_load_b128) per fragment.
// ---------------------------------------------------------------------------
__device__ __forceinline__ v16h load_a_frag(const _Float16* __restrict__ A,
                                            int lda, int row_base, int kb) {
  const int lane = threadIdx.x & 31;
  const int row  = row_base + (lane & 15);
  const int sel  = (lane >> 4) << 3;           // 0 or 8
  const _Float16* p0 = A + (size_t)row * lda + kb + sel;
  const _Float16* p1 = p0 + 16;
  v16h a;
#pragma unroll
  for (int j = 0; j < 8; ++j) { a[j] = p0[j]; a[8 + j] = p1[j]; }
  return a;
}

// Swizzled-B fragment: base = ((kt*ntiles + nt)*32 + lane)*16
__device__ __forceinline__ v16h load_b_sw(const _Float16* __restrict__ Bsw,
                                          int ntiles, int kt, int nt) {
  const int lane = threadIdx.x & 31;
  const _Float16* p = Bsw + ((size_t)((kt * ntiles + nt) * 32 + lane) << 4);
  v16h b;
#pragma unroll
  for (int j = 0; j < 16; ++j) b[j] = p[j];
  return b;
}

// map flat swizzled index -> (k, n) source coordinates
__device__ __forceinline__ void sw_coords(int idx, int ntiles, int& k, int& n) {
  int j    = idx & 15;
  int lane = (idx >> 4) & 31;
  int tile = idx >> 9;                 // / (32*16)
  int nt   = tile % ntiles;
  int kt   = tile / ntiles;
  int sel  = (lane >> 4) << 3;
  int kin  = (j < 8) ? (sel + j) : (16 + sel + (j - 8));
  k = kt * 32 + kin;
  n = nt * 16 + (lane & 15);
}

__device__ __forceinline__ v8f wmma32(v16h a, v16h b, v8f c) {
  return __builtin_amdgcn_wmma_f32_16x16x32_f16(false, a, false, b,
                                                (short)0, c, false, false);
}

__device__ __forceinline__ float fatomic_add(float* p, float v) {
  return __hip_atomic_fetch_add(p, v, __ATOMIC_RELAXED, __HIP_MEMORY_SCOPE_AGENT);
}

// ---------------------------------------------------------------------------
// Degree / normalization
// ---------------------------------------------------------------------------
__global__ void k_deg_zero(float* deg) {
  int i = blockIdx.x * blockDim.x + threadIdx.x;
  if (i < NN) deg[i] = 0.f;
}
__global__ void k_deg_count(const long long* __restrict__ ei, float* deg) {
  int e = blockIdx.x * blockDim.x + threadIdx.x;
  if (e >= NE) return;
  fatomic_add(&deg[ei[NE + e]], 1.0f);
}
__global__ void k_dinv(float* deg) {
  int i = blockIdx.x * blockDim.x + threadIdx.x;
  if (i < NN) deg[i] = rsqrtf(deg[i] + 1.0f);
}

// ---------------------------------------------------------------------------
// Stage 1: x[:,13:16,:] @ W1  (K=4, scalar VALU)
// h1pre layout: plane-major [tt][n][c], tt=0..2 <-> t=13..15
// ---------------------------------------------------------------------------
__global__ void k_lin1(const float* __restrict__ x, const float* __restrict__ W1,
                       float* __restrict__ h1pre) {
  int idx = blockIdx.x * blockDim.x + threadIdx.x;
  if (idx >= NN * H1) return;
  int n = idx / H1, c = idx % H1;
  const float* xp = x + (size_t)n * SEQ * FIN + 13 * FIN;
#pragma unroll
  for (int tt = 0; tt < 3; ++tt) {
    float a = 0.f;
#pragma unroll
    for (int f = 0; f < FIN; ++f) a += xp[tt * FIN + f] * W1[f * H1 + c];
    h1pre[(size_t)tt * NN * H1 + (size_t)n * H1 + c] = a;
  }
}

// GCN aggregation: self term init, edge scatter (HW f32 atomics), bias+relu+cvt
__global__ void k_agg1_init(const float* __restrict__ h1pre,
                            const float* __restrict__ dinv, float* agg1) {
  int idx = blockIdx.x * blockDim.x + threadIdx.x;
  if (idx >= 3 * NN * H1) return;
  int n = (idx / H1) % NN;
  float di = dinv[n];
  agg1[idx] = h1pre[idx] * di * di;
}
__global__ void k_agg1_scatter(const long long* __restrict__ ei,
                               const float* __restrict__ h1pre,
                               const float* __restrict__ dinv, float* agg1) {
  int e = blockIdx.x, c = threadIdx.x;                 // blockDim = 64
  long long s = ei[e], d = ei[NE + e];
  float norm = dinv[s] * dinv[d];
#pragma unroll
  for (int tt = 0; tt < 3; ++tt)
    fatomic_add(&agg1[(size_t)tt * NN * H1 + (size_t)d * H1 + c],
                h1pre[(size_t)tt * NN * H1 + (size_t)s * H1 + c] * norm);
}
__global__ void k_relu1_cvt(const float* __restrict__ agg1,
                            const float* __restrict__ b1, _Float16* h1f16) {
  int idx = blockIdx.x * blockDim.x + threadIdx.x;
  if (idx >= 3 * NN * H1) return;
  float v = agg1[idx] + b1[idx % H1];
  h1f16[idx] = (_Float16)(v > 0.f ? v : 0.f);
}

// ---------------------------------------------------------------------------
// Weight prep: f32 -> f16, directly into WMMA fragment-swizzled layout
// ---------------------------------------------------------------------------
__global__ void k_prep_k1(const float* __restrict__ K1, _Float16* k1t) {
  int idx = blockIdx.x * blockDim.x + threadIdx.x;
  if (idx >= 3 * H1 * H1) return;
  int dt = idx / (H1 * H1), rem = idx % (H1 * H1);
  int k, n; sw_coords(rem, H1 / 16, k, n);
  k1t[idx] = (_Float16)K1[(size_t)n * H1 * 3 + k * 3 + dt];  // K1[o=n][i=k][dt]
}
__global__ void k_prep_w2(const float* __restrict__ W2, _Float16* w) {
  int idx = blockIdx.x * blockDim.x + threadIdx.x;
  if (idx >= H1 * H2) return;
  int k, n; sw_coords(idx, H2 / 16, k, n);
  w[idx] = (_Float16)W2[(size_t)k * H2 + n];
}
__global__ void k_prep_k2(const float* __restrict__ K2, _Float16* k2t) {
  int idx = blockIdx.x * blockDim.x + threadIdx.x;
  if (idx >= 2 * H2 * H2) return;
  int dt = idx / (H2 * H2), rem = idx % (H2 * H2);
  int k, n; sw_coords(rem, H2 / 16, k, n);
  k2t[idx] = (_Float16)K2[(size_t)n * H2 * 3 + k * 3 + dt];
}
__global__ void k_prep_f1(const float* __restrict__ F1, _Float16* w) {
  int idx = blockIdx.x * blockDim.x + threadIdx.x;
  if (idx >= H2 * H1) return;
  int k, n; sw_coords(idx, H1 / 16, k, n);
  w[idx] = (_Float16)F1[(size_t)k * H1 + n];
}

// ---------------------------------------------------------------------------
// tconv1 as node-batched GEMM (WMMA): out t in {14,15} -> planes to in {0,1}
//   Y[to] = sum_dt H1[to+dt] @ K1t[dt]   (to==1 drops dt=2: t=16 zero pad)
// ---------------------------------------------------------------------------
__global__ void k_tconv1(const _Float16* __restrict__ h1f16,
                         const _Float16* __restrict__ k1t,
                         const float* __restrict__ kb1,
                         _Float16* __restrict__ t1f16) {
  const int wave = blockIdx.x * (blockDim.x >> 5) + (threadIdx.x >> 5);
  const int to = blockIdx.y;
  if (wave >= NN / 16) return;
  const int row = wave * 16;
  v8f acc[4] = {};
  const int ndt = (to == 0) ? 3 : 2;
  for (int dt = 0; dt < ndt; ++dt) {
    const _Float16* A = h1f16 + (size_t)(to + dt) * NN * H1;
    const _Float16* B = k1t + (size_t)dt * H1 * H1;
#pragma unroll
    for (int kt = 0; kt < H1 / 32; ++kt) {
      v16h a = load_a_frag(A, H1, row, kt * 32);
#pragma unroll
      for (int nt = 0; nt < 4; ++nt)
        acc[nt] = wmma32(a, load_b_sw(B, 4, kt, nt), acc[nt]);
    }
  }
  const int lane = threadIdx.x & 31;
  const int colo = lane & 15, mh = (lane >> 4) << 3;
#pragma unroll
  for (int nt = 0; nt < 4; ++nt)
#pragma unroll
    for (int j = 0; j < 8; ++j) {
      int r = row + mh + j, c = nt * 16 + colo;
      float v = acc[nt][j] + kb1[c];
      t1f16[(size_t)to * NN * H1 + (size_t)r * H1 + c] = (_Float16)(v > 0.f ? v : 0.f);
    }
}

// lin2: (2*NN x 64) @ (64 x 128) WMMA -> h2pre f32 (plane-major [to][n][o])
__global__ void k_lin2(const _Float16* __restrict__ t1f16,
                       const _Float16* __restrict__ w2f16,
                       float* __restrict__ h2pre) {
  const int wave = blockIdx.x * (blockDim.x >> 5) + (threadIdx.x >> 5);
  if (wave >= (2 * NN) / 16) return;
  const int row = wave * 16;
  v8f acc[8] = {};
#pragma unroll
  for (int kt = 0; kt < H1 / 32; ++kt) {
    v16h a = load_a_frag(t1f16, H1, row, kt * 32);
#pragma unroll
    for (int nt = 0; nt < 8; ++nt)
      acc[nt] = wmma32(a, load_b_sw(w2f16, 8, kt, nt), acc[nt]);
  }
  const int lane = threadIdx.x & 31;
  const int colo = lane & 15, mh = (lane >> 4) << 3;
#pragma unroll
  for (int nt = 0; nt < 8; ++nt)
#pragma unroll
    for (int j = 0; j < 8; ++j)
      h2pre[(size_t)(row + mh + j) * H2 + nt * 16 + colo] = acc[nt][j];
}

__global__ void k_agg2_init(const float* __restrict__ h2pre,
                            const float* __restrict__ dinv, float* agg2) {
  int idx = blockIdx.x * blockDim.x + threadIdx.x;
  if (idx >= 2 * NN * H2) return;
  int n = (idx / H2) % NN;
  float di = dinv[n];
  agg2[idx] = h2pre[idx] * di * di;
}
__global__ void k_agg2_scatter(const long long* __restrict__ ei,
                               const float* __restrict__ h2pre,
                               const float* __restrict__ dinv, float* agg2) {
  int e = blockIdx.x, c = threadIdx.x;                 // blockDim = 128
  long long s = ei[e], d = ei[NE + e];
  float norm = dinv[s] * dinv[d];
#pragma unroll
  for (int to = 0; to < 2; ++to)
    fatomic_add(&agg2[(size_t)to * NN * H2 + (size_t)d * H2 + c],
                h2pre[(size_t)to * NN * H2 + (size_t)s * H2 + c] * norm);
}
__global__ void k_relu2_cvt(const float* __restrict__ agg2,
                            const float* __restrict__ b2, _Float16* h2f16) {
  int idx = blockIdx.x * blockDim.x + threadIdx.x;
  if (idx >= 2 * NN * H2) return;
  float v = agg2[idx] + b2[idx % H2];
  h2f16[idx] = (_Float16)(v > 0.f ? v : 0.f);
}

// tconv2: only t=15 output.  last = relu(sum_{dt=0,1} H2[dt] @ K2t[dt] + kb2)
__global__ void k_tconv2(const _Float16* __restrict__ h2f16,
                         const _Float16* __restrict__ k2t,
                         const float* __restrict__ kb2,
                         _Float16* __restrict__ lastf16) {
  const int wave = blockIdx.x * (blockDim.x >> 5) + (threadIdx.x >> 5);
  if (wave >= NN / 16) return;
  const int row = wave * 16;
  v8f acc[8] = {};
  for (int dt = 0; dt < 2; ++dt) {
    const _Float16* A = h2f16 + (size_t)dt * NN * H2;
    const _Float16* B = k2t + (size_t)dt * H2 * H2;
#pragma unroll
    for (int kt = 0; kt < H2 / 32; ++kt) {
      v16h a = load_a_frag(A, H2, row, kt * 32);
#pragma unroll
      for (int nt = 0; nt < 8; ++nt)
        acc[nt] = wmma32(a, load_b_sw(B, 8, kt, nt), acc[nt]);
    }
  }
  const int lane = threadIdx.x & 31;
  const int colo = lane & 15, mh = (lane >> 4) << 3;
#pragma unroll
  for (int nt = 0; nt < 8; ++nt)
#pragma unroll
    for (int j = 0; j < 8; ++j) {
      int r = row + mh + j, c = nt * 16 + colo;
      float v = acc[nt][j] + kb2[c];
      lastf16[(size_t)r * H2 + c] = (_Float16)(v > 0.f ? v : 0.f);
    }
}

// head: z = relu(last @ F1 + fb1)   (10000x128 @ 128x64, WMMA)
__global__ void k_headz(const _Float16* __restrict__ lastf16,
                        const _Float16* __restrict__ f1f16,
                        const float* __restrict__ fb1, float* __restrict__ zf32) {
  const int wave = blockIdx.x * (blockDim.x >> 5) + (threadIdx.x >> 5);
  if (wave >= NN / 16) return;
  const int row = wave * 16;
  v8f acc[4] = {};
#pragma unroll
  for (int kt = 0; kt < H2 / 32; ++kt) {
    v16h a = load_a_frag(lastf16, H2, row, kt * 32);
#pragma unroll
    for (int nt = 0; nt < 4; ++nt)
      acc[nt] = wmma32(a, load_b_sw(f1f16, 4, kt, nt), acc[nt]);
  }
  const int lane = threadIdx.x & 31;
  const int colo = lane & 15, mh = (lane >> 4) << 3;
#pragma unroll
  for (int nt = 0; nt < 4; ++nt)
#pragma unroll
    for (int j = 0; j < 8; ++j) {
      int r = row + mh + j, c = nt * 16 + colo;
      float v = acc[nt][j] + fb1[c];
      zf32[(size_t)r * H1 + c] = v > 0.f ? v : 0.f;
    }
}

// pred = softplus(z @ F2 + fb2)   (N=12, scalar)
__global__ void k_headout(const float* __restrict__ zf32,
                          const float* __restrict__ F2,
                          const float* __restrict__ fb2, float* __restrict__ out) {
  int idx = blockIdx.x * blockDim.x + threadIdx.x;
  if (idx >= NN * HOR * FIN) return;
  int n = idx / (HOR * FIN), k = idx % (HOR * FIN);
  float acc = fb2[k];
#pragma unroll 8
  for (int j = 0; j < H1; ++j) acc += zf32[(size_t)n * H1 + j] * F2[j * (HOR * FIN) + k];
  out[idx] = (acc > 20.f) ? acc : log1pf(__expf(acc));
}

// ---------------------------------------------------------------------------
extern "C" void kernel_launch(void* const* d_in, const int* in_sizes, int n_in,
                              void* d_out, int out_size, void* d_ws, size_t ws_size,
                              hipStream_t stream) {
  (void)in_sizes; (void)n_in; (void)out_size; (void)ws_size;
  const float*     x   = (const float*)d_in[0];
  const long long* ei  = (const long long*)d_in[1];
  const float*     W1  = (const float*)d_in[2];
  const float*     b1  = (const float*)d_in[3];
  const float*     K1  = (const float*)d_in[4];
  const float*     kb1 = (const float*)d_in[5];
  const float*     W2  = (const float*)d_in[6];
  const float*     b2  = (const float*)d_in[7];
  const float*     K2  = (const float*)d_in[8];
  const float*     kb2 = (const float*)d_in[9];
  const float*     F1  = (const float*)d_in[10];
  const float*     fb1 = (const float*)d_in[11];
  const float*     F2  = (const float*)d_in[12];
  const float*     fb2 = (const float*)d_in[13];

  unsigned char* ws = (unsigned char*)d_ws;
  size_t off = 0;
  auto alloc = [&](size_t bytes) {
    size_t o = off; off = (off + bytes + 255) & ~(size_t)255; return o;
  };
  float*     dinv    = (float*)(ws + alloc((size_t)NN * 4));
  float*     h1pre   = (float*)(ws + alloc(3ull * NN * H1 * 4));
  float*     agg1    = (float*)(ws + alloc(3ull * NN * H1 * 4));
  _Float16*  h1f16   = (_Float16*)(ws + alloc(3ull * NN * H1 * 2));
  _Float16*  k1t     = (_Float16*)(ws + alloc(3ull * H1 * H1 * 2));
  _Float16*  t1f16   = (_Float16*)(ws + alloc(2ull * NN * H1 * 2));
  _Float16*  w2f16   = (_Float16*)(ws + alloc((size_t)H1 * H2 * 2));
  float*     h2pre   = (float*)(ws + alloc(2ull * NN * H2 * 4));
  float*     agg2    = (float*)(ws + alloc(2ull * NN * H2 * 4));
  _Float16*  h2f16   = (_Float16*)(ws + alloc(2ull * NN * H2 * 2));
  _Float16*  k2t     = (_Float16*)(ws + alloc(2ull * H2 * H2 * 2));
  _Float16*  lastf16 = (_Float16*)(ws + alloc((size_t)NN * H2 * 2));
  _Float16*  f1f16   = (_Float16*)(ws + alloc((size_t)H2 * H1 * 2));
  float*     zf32    = (float*)(ws + alloc((size_t)NN * H1 * 4));

  const int B = 256;
  k_deg_zero <<<(NN + B - 1) / B, B, 0, stream>>>(dinv);
  k_deg_count<<<(NE + B - 1) / B, B, 0, stream>>>(ei, dinv);
  k_dinv     <<<(NN + B - 1) / B, B, 0, stream>>>(dinv);

  k_lin1     <<<(NN * H1 + B - 1) / B, B, 0, stream>>>(x, W1, h1pre);
  k_agg1_init<<<(3 * NN * H1 + B - 1) / B, B, 0, stream>>>(h1pre, dinv, agg1);
  k_agg1_scatter<<<NE, H1, 0, stream>>>(ei, h1pre, dinv, agg1);
  k_relu1_cvt<<<(3 * NN * H1 + B - 1) / B, B, 0, stream>>>(agg1, b1, h1f16);

  k_prep_k1<<<(3 * H1 * H1 + B - 1) / B, B, 0, stream>>>(K1, k1t);
  dim3 g1((NN / 16 + 3) / 4, 2);
  k_tconv1<<<g1, 128, 0, stream>>>(h1f16, k1t, kb1, t1f16);

  k_prep_w2<<<(H1 * H2 + B - 1) / B, B, 0, stream>>>(W2, w2f16);
  k_lin2<<<(2 * NN / 16 + 3) / 4, 128, 0, stream>>>(t1f16, w2f16, h2pre);

  k_agg2_init<<<(2 * NN * H2 + B - 1) / B, B, 0, stream>>>(h2pre, dinv, agg2);
  k_agg2_scatter<<<NE, H2, 0, stream>>>(ei, h2pre, dinv, agg2);
  k_relu2_cvt<<<(2 * NN * H2 + B - 1) / B, B, 0, stream>>>(agg2, b2, h2f16);

  k_prep_k2<<<(2 * H2 * H2 + B - 1) / B, B, 0, stream>>>(K2, k2t);
  k_tconv2<<<(NN / 16 + 3) / 4, 128, 0, stream>>>(h2f16, k2t, kb2, lastf16);

  k_prep_f1<<<(H2 * H1 + B - 1) / B, B, 0, stream>>>(F1, f1f16);
  k_headz<<<(NN / 16 + 3) / 4, 128, 0, stream>>>(lastf16, f1f16, fb1, zf32);
  k_headout<<<(NN * HOR * FIN + B - 1) / B, B, 0, stream>>>(zf32, F2, fb2, (float*)d_out);
}
